// WaveNet_10299331576122
// MI455X (gfx1250) — matrix operand order repaired and checked by reading, not compiled, not run on hardware
//
#include <hip/hip_runtime.h>
#include <hip/hip_bf16.h>

typedef __attribute__((ext_vector_type(16))) _Float16 v16h;
typedef __attribute__((ext_vector_type(8)))  _Float16 v8h;
typedef __attribute__((ext_vector_type(8)))  float    v8f;

#define C16 16
#define TT  32768
#define BB  8
#define LL  16

__device__ __forceinline__ v8f wmma_f16(v16h a, v16h b, v8f c) {
    return __builtin_amdgcn_wmma_f32_16x16x32_f16(false, a, false, b, (short)0, c, false, false);
}

// branch-free transcendental gate (v_tanh_f32 / v_exp_f32 / v_rcp_f32 on CDNA5)
__device__ __forceinline__ float fast_sigmoid(float x) {
    return __builtin_amdgcn_rcpf(1.f + __builtin_amdgcn_exp2f(x * -1.44269504089f));
}
__device__ __forceinline__ float fast_tanh(float x) {
#if __has_builtin(__builtin_amdgcn_tanhf)
    return __builtin_amdgcn_tanhf(x);
#else
    float e = __builtin_amdgcn_exp2f(x * -2.88539008178f);   // exp(-2x)
    return __builtin_amdgcn_rcpf(1.f + e) * 2.f - 1.f;
#endif
}

// ---------------------------------------------------------------------------
// Prep: input 1x1 conv (1 -> C) into f16 channel-last activations [B][T][C],
// init out = mix_b.
// ---------------------------------------------------------------------------
__global__ void wavenet_prep(const float* __restrict__ x,
                             const float* __restrict__ in_w,
                             const float* __restrict__ in_b,
                             const float* __restrict__ mix_b,
                             _Float16* __restrict__ act,
                             float* __restrict__ out)
{
    int idx = blockIdx.x * blockDim.x + threadIdx.x;   // over B*T*C (C fastest)
    int c = idx & 15;
    int t = (idx >> 4) & (TT - 1);
    int b = idx >> 19;
    float xv = x[(size_t)b * TT + t];
    act[idx] = (_Float16)(xv * in_w[c] + in_b[c]);
    if (c == 0) out[(size_t)b * TT + t] = mix_b[0];
}

// ---------------------------------------------------------------------------
// One-shot weight packer: lay out f16 A-matrices per lane with
// contiguous k = 16*half + s (same order as B), and bias/mix vectors per the f32 C/D layout.
//   wpack: [L][5][32 lanes][16 halfs]  (aF0, aF1, aG0, aG1, aR)
//   bpack: [L][4][32 lanes][8 floats]  (biasF, biasG, biasR, mixw)
// ---------------------------------------------------------------------------
__global__ void wavenet_pack(const float* __restrict__ hidW,  // [L][2C][C][3]
                             const float* __restrict__ hidB,  // [L][2C]
                             const float* __restrict__ resW,  // [L][C][C]
                             const float* __restrict__ resB,  // [L][C]
                             const float* __restrict__ mixW,  // [C*L]
                             _Float16* __restrict__ wpack,
                             float*    __restrict__ bpack)
{
    int idx = blockIdx.x * blockDim.x + threadIdx.x;   // 0..511 = (layer, lane)
    int lane  = idx & 31;
    int layer = idx >> 5;
    int hi = lane >> 4;
    int mn = lane & 15;

    _Float16* wa = wpack + (size_t)layer * 5 * 32 * 16;
#pragma unroll
    for (int j = 0; j < 8; ++j) {
#pragma unroll
        for (int h = 0; h < 2; ++h) {
            int e  = 2 * j + h;
            int ql = (hi ? 16 : 0) + e; // 0..31
            int k = ql >> 4, cin = ql & 15;
            float f0 = hidW[(((size_t)layer * 32 +      mn) * 16 + cin) * 3 + k];
            float g0 = hidW[(((size_t)layer * 32 + 16 + mn) * 16 + cin) * 3 + k];
            float f1 = (ql < 16) ? hidW[(((size_t)layer * 32 +      mn) * 16 + ql) * 3 + 2] : 0.f;
            float g1 = (ql < 16) ? hidW[(((size_t)layer * 32 + 16 + mn) * 16 + ql) * 3 + 2] : 0.f;
            float rw = (ql < 16) ? resW[((size_t)layer * 16 + mn) * 16 + ql] : 0.f;
            wa[(0 * 32 + lane) * 16 + e] = (_Float16)f0;
            wa[(1 * 32 + lane) * 16 + e] = (_Float16)f1;
            wa[(2 * 32 + lane) * 16 + e] = (_Float16)g0;
            wa[(3 * 32 + lane) * 16 + e] = (_Float16)g1;
            wa[(4 * 32 + lane) * 16 + e] = (_Float16)rw;
        }
    }
    float* bp = bpack + (size_t)layer * 4 * 32 * 8;
#pragma unroll
    for (int r = 0; r < 8; ++r) {
        int m = r + (hi ? 8 : 0);
        bp[(0 * 32 + lane) * 8 + r] = hidB[(size_t)layer * 32 + m];
        bp[(1 * 32 + lane) * 8 + r] = hidB[(size_t)layer * 32 + 16 + m];
        bp[(2 * 32 + lane) * 8 + r] = resB[(size_t)layer * 16 + m];
        bp[(3 * 32 + lane) * 8 + r] = mixW[(size_t)layer * 16 + m];
    }
}

// ---------------------------------------------------------------------------
// One WaveNet layer. One wave handles 4 consecutive (16-timestep) tiles of
// one batch row. Activations are [B][T][C] f16 so every matrix fragment is a
// contiguous vector load.
// ---------------------------------------------------------------------------
__global__ void __launch_bounds__(256, 2) wavenet_layer(
    const _Float16* __restrict__ actIn,   // [B][T][C] f16
    _Float16*       __restrict__ actOut,  // [B][T][C] f16
    const _Float16* __restrict__ wpack,   // [L][5][32][16]
    const float*    __restrict__ bpack,   // [L][4][32][8]
    float*          __restrict__ out,     // [B][T] running skip-mix accum
    int layer, int dil)
{
    __shared__ _Float16 ldsG[8][C16 * 16];

    const int lane = threadIdx.x & 31;
    const int wv   = threadIdx.x >> 5;
    const int hi   = lane >> 4;
    const int mn   = lane & 15;             // M for A, N for B/C/D
    const int g    = blockIdx.x * 8 + wv;   // 0..4095 tile-groups
    const int b    = g >> 9;                // 512 groups per batch row
    const int tb   = (g & 511) << 6;        // 4 tiles * 16 timesteps

    // ---- preplaced weights / biases: pure vector loads ----------------------
    const v16h* wp = (const v16h*)wpack + (size_t)layer * 5 * 32;
    const v16h aF0 = wp[0 * 32 + lane];
    const v16h aF1 = wp[1 * 32 + lane];
    const v16h aG0 = wp[2 * 32 + lane];
    const v16h aG1 = wp[3 * 32 + lane];
    const v16h aR  = wp[4 * 32 + lane];
    const v8f* bp = (const v8f*)bpack + (size_t)layer * 4 * 32;
    const v8f biasF = bp[0 * 32 + lane];
    const v8f biasG = bp[1 * 32 + lane];
    const v8f biasR = bp[2 * 32 + lane];
    const v8f mixv  = bp[3 * 32 + lane];

    const _Float16* actB  = actIn  + (size_t)b * TT * C16;
    _Float16*       actOB = actOut + (size_t)b * TT * C16;
    const v16h z16 = {};

#pragma unroll
    for (int ti = 0; ti < 4; ++ti) {
        const int tn = tb + ti * 16 + mn;   // this lane's timestep (N index)

        // B chunk0: lanes 0-15 hold K=cin @ tap k=0 (t-2d); lanes 16-31 k=1 (t-d)
        int tk = tn - (hi ? dil : 2 * dil);
        v16h bh0 = (tk >= 0) ? *(const v16h*)(actB + (size_t)tk * C16) : z16;
        // B chunk1: lanes 0-15 tap k=2 (t); lanes 16-31 are K=48..63 zero-pad
        v16h bh1 = hi ? z16 : *(const v16h*)(actB + (size_t)tn * C16);

        v8f cF = biasF, cG = biasG;
        cF = wmma_f16(aF0, bh0, cF);
        cF = wmma_f16(aF1, bh1, cF);
        cG = wmma_f16(aG0, bh0, cG);
        cG = wmma_f16(aG1, bh1, cG);

        // ---- gate + skip/mix partial; stage gated into LDS as [n][cin] ------
        v8h gh;
        float part = 0.f;
#pragma unroll
        for (int r = 0; r < 8; ++r) {
            float gv = fast_tanh(cF[r]) * fast_sigmoid(cG[r]);
            gh[r] = (_Float16)gv;
            part += mixv[r] * gv;
        }
        // lane holds rows m = r+8*hi of column n -> contiguous 16B store
        *(v8h*)(&ldsG[wv][mn * 16 + hi * 8]) = gh;
        asm volatile("s_wait_dscnt 0" ::: "memory");   // wave-lockstep LDS sync
        part += __shfl_xor(part, 16);
        if (!hi) out[(size_t)b * TT + tn] += part;

        // ---- residual 1x1 conv: B from LDS (K=cin, zero-pad upper half) -----
        v16h bR = hi ? z16 : *(const v16h*)(&ldsG[wv][mn * 16]);
        v8h xi = *(const v8h*)(actB + (size_t)tn * C16 + hi * 8);
        v8f cR = biasR;
#pragma unroll
        for (int r = 0; r < 8; ++r) cR[r] += (float)xi[r];
        cR = wmma_f16(aR, bR, cR);

        v8h ro;
#pragma unroll
        for (int r = 0; r < 8; ++r) ro[r] = (_Float16)cR[r];
        *(v8h*)(actOB + (size_t)tn * C16 + hi * 8) = ro;
    }
}

// ---------------------------------------------------------------------------
extern "C" void kernel_launch(void* const* d_in, const int* in_sizes, int n_in,
                              void* d_out, int out_size, void* d_ws, size_t ws_size,
                              hipStream_t stream)
{
    const float* x     = (const float*)d_in[0];
    const float* in_w  = (const float*)d_in[1];
    const float* in_b  = (const float*)d_in[2];
    const float* hid_w = (const float*)d_in[3];
    const float* hid_b = (const float*)d_in[4];
    const float* res_w = (const float*)d_in[5];
    const float* res_b = (const float*)d_in[6];
    const float* mix_w = (const float*)d_in[7];
    const float* mix_b = (const float*)d_in[8];
    float* out = (float*)d_out;

    const size_t actElems = (size_t)BB * TT * C16;
    _Float16* buf0  = (_Float16*)d_ws;
    _Float16* buf1  = buf0 + actElems;                 // 2 x 8 MB f16 ping-pong
    _Float16* wpack = buf1 + actElems;                 // 80 KB packed A-mats
    float*    bpack = (float*)(wpack + (size_t)LL * 5 * 32 * 16);  // 64 KB

    wavenet_pack<<<2, 256, 0, stream>>>(hid_w, hid_b, res_w, res_b, mix_w,
                                        wpack, bpack);
    wavenet_prep<<<(int)(actElems / 256), 256, 0, stream>>>(x, in_w, in_b, mix_b,
                                                            buf0, out);

    _Float16* cur = buf0;
    _Float16* nxt = buf1;
    for (int i = 0; i < LL; ++i) {
        int dil = 1 << (i & 7);                        // [1..128] repeated twice
        wavenet_layer<<<512, 256, 0, stream>>>(cur, nxt, wpack, bpack, out, i, dil);
        _Float16* tmp = cur; cur = nxt; nxt = tmp;
    }
}
